// GeeArrYou_70128226009391
// MI455X (gfx1250) — matrix operand order repaired and checked by reading, not compiled, not run on hardware
//
#include <hip/hip_runtime.h>
#include <hip/hip_bf16.h>

#define BB   32
#define LL   512
#define IN_F 512
#define HH   1024
#define NLAY 4
#define G3   (3 * HH)  // 3072

typedef __attribute__((ext_vector_type(16))) __bf16 v16bf;
typedef __attribute__((ext_vector_type(8)))  float  v8f;
typedef __attribute__((ext_vector_type(4)))  unsigned int u32x4;
typedef __attribute__((ext_vector_type(8)))  int          i32x8;
typedef __attribute__((ext_vector_type(4)))  int          i32x4;

union BF16Frag { uint4 q[2]; v16bf v; };

__device__ __forceinline__ unsigned short f2bf(float f) {
    unsigned int u = __float_as_uint(f);
    unsigned int r = u + 0x7FFFu + ((u >> 16) & 1u);  // round-to-nearest-even
    return (unsigned short)(r >> 16);
}
__device__ __forceinline__ float bf2f(unsigned short h) {
    return __uint_as_float(((unsigned int)h) << 16);
}
__device__ __forceinline__ float sigmoidf_(float x) { return 1.0f / (1.0f + __expf(-x)); }

// A-fragment (16x32 bf16, M x K): lane l holds row m=l&15, K chunks [k0..k0+7],[k0+16..k0+23], k0=(l>>4)*8
__device__ __forceinline__ v16bf load_a_frag(const unsigned short* base, int ld, int lane) {
    int m = lane & 15, k0 = (lane >> 4) * 8;
    const unsigned short* p = base + (size_t)m * ld + k0;
    BF16Frag f;
    f.q[0] = *(const uint4*)(p);
    f.q[1] = *(const uint4*)(p + 16);
    return f.v;
}
// B-fragment (32x16 bf16, K x N): lane l holds column n=l&15, K chunks [k0..k0+7],[k0+16..k0+23], k0=(l>>4)*8
__device__ __forceinline__ v16bf load_b_frag(const unsigned short* base, int ld, int lane) {
    int n = lane & 15, k0 = (lane >> 4) * 8;
    const unsigned short* p = base + (size_t)n * ld + k0;
    BF16Frag f;
    f.q[0] = *(const uint4*)(p);
    f.q[1] = *(const uint4*)(p + 16);
    return f.v;
}

// ---------------- casts / setup ----------------
__global__ void k_cast_bf16(const float* __restrict__ in, unsigned short* __restrict__ out, size_t n) {
    size_t i = (size_t)blockIdx.x * blockDim.x + threadIdx.x;
    size_t s = (size_t)gridDim.x * blockDim.x;
    for (; i < n; i += s) out[i] = f2bf(in[i]);
}

// x: [B][L][IN] f32  ->  Xb[(t+1)*B*IN + b*IN + f] bf16  (time-major, slice 0 unused)
__global__ void k_cast_x(const float* __restrict__ x, unsigned short* __restrict__ Xb) {
    size_t total = (size_t)BB * LL * IN_F;
    size_t i = (size_t)blockIdx.x * blockDim.x + threadIdx.x;
    size_t s = (size_t)gridDim.x * blockDim.x;
    for (; i < total; i += s) {
        size_t f = i % IN_F;
        size_t t = (i / IN_F) % LL;
        size_t b = i / ((size_t)IN_F * LL);
        Xb[(t + 1) * (size_t)BB * IN_F + b * IN_F + f] = f2bf(x[i]);
    }
}

// biasC = b_ih + (gate r,z ? b_hh : 0)
__global__ void k_biasc(const float* __restrict__ bih, const float* __restrict__ bhh,
                        float* __restrict__ biasC) {
    int i = blockIdx.x * blockDim.x + threadIdx.x;
    if (i < NLAY * G3) {
        int g = i % G3;
        biasC[i] = bih[i] + (g < 2 * HH ? bhh[i] : 0.0f);
    }
}

__global__ void k_zero(unsigned int* __restrict__ flags, unsigned int* __restrict__ h0,
                       int nflags, int nh0) {
    int i = blockIdx.x * blockDim.x + threadIdx.x;
    if (i < nflags) flags[i] = 0u;
    if (i < nh0)    h0[i]   = 0u;
}

// ---------------- input projection GEMM: XP[m][g] = X[m][:]*W[g][:] + biasC[g] ----------------
// grid = (G3/256, (L*B)/32), block = 128 (4 waves).
// Each wave: 32(M) x 64(N): 2 A-fragments shared across 4 B-fragments -> 8 WMMAs per k-step,
// halving L2 weight traffic vs a single M-tile.
__global__ __launch_bounds__(128) void k_xproj(
    const unsigned short* __restrict__ X,   // [L*B][F] row-major (pre-offset to time slice 1)
    const unsigned short* __restrict__ W,   // [G3][F] row-major
    const float* __restrict__ biasC,        // [G3]
    float* __restrict__ XP, int F)          // [L*B][G3]
{
    const int lane  = threadIdx.x & 31;
    const int wave  = threadIdx.x >> 5;
    const int mbase = blockIdx.y * 32;
    const int gbase = blockIdx.x * 256 + wave * 64;
    const unsigned short* Arow0 = X + (size_t)mbase * F;
    const unsigned short* Arow1 = X + (size_t)(mbase + 16) * F;

    v8f a00 = {}, a01 = {}, a02 = {}, a03 = {};
    v8f a10 = {}, a11 = {}, a12 = {}, a13 = {};
    const int KT = F >> 5;
    for (int kk = 0; kk < KT; ++kk) {
        v16bf a0 = load_a_frag(Arow0 + kk * 32, F, lane);
        v16bf a1 = load_a_frag(Arow1 + kk * 32, F, lane);
        v16bf b0 = load_b_frag(W + (size_t)(gbase +  0) * F + kk * 32, F, lane);
        a00 = __builtin_amdgcn_wmma_f32_16x16x32_bf16(false, a0, false, b0, (short)0, a00, false, false);
        a10 = __builtin_amdgcn_wmma_f32_16x16x32_bf16(false, a1, false, b0, (short)0, a10, false, false);
        v16bf b1 = load_b_frag(W + (size_t)(gbase + 16) * F + kk * 32, F, lane);
        a01 = __builtin_amdgcn_wmma_f32_16x16x32_bf16(false, a0, false, b1, (short)0, a01, false, false);
        a11 = __builtin_amdgcn_wmma_f32_16x16x32_bf16(false, a1, false, b1, (short)0, a11, false, false);
        v16bf b2 = load_b_frag(W + (size_t)(gbase + 32) * F + kk * 32, F, lane);
        a02 = __builtin_amdgcn_wmma_f32_16x16x32_bf16(false, a0, false, b2, (short)0, a02, false, false);
        a12 = __builtin_amdgcn_wmma_f32_16x16x32_bf16(false, a1, false, b2, (short)0, a12, false, false);
        v16bf b3 = load_b_frag(W + (size_t)(gbase + 48) * F + kk * 32, F, lane);
        a03 = __builtin_amdgcn_wmma_f32_16x16x32_bf16(false, a0, false, b3, (short)0, a03, false, false);
        a13 = __builtin_amdgcn_wmma_f32_16x16x32_bf16(false, a1, false, b3, (short)0, a13, false, false);
    }
    const int mrow = (lane >> 4) * 8;
    const int n    = lane & 15;
    const float c0 = biasC[gbase +  0 + n];
    const float c1 = biasC[gbase + 16 + n];
    const float c2 = biasC[gbase + 32 + n];
    const float c3 = biasC[gbase + 48 + n];
    #pragma unroll
    for (int v = 0; v < 8; ++v) {
        float* o0 = XP + (size_t)(mbase + mrow + v) * G3 + gbase + n;
        o0[0] = a00[v] + c0; o0[16] = a01[v] + c1; o0[32] = a02[v] + c2; o0[48] = a03[v] + c3;
        float* o1 = XP + (size_t)(mbase + 16 + mrow + v) * G3 + gbase + n;
        o1[0] = a10[v] + c0; o1[16] = a11[v] + c1; o1[32] = a12[v] + c2; o1[48] = a13[v] + c3;
    }
}

// ---------------- persistent recurrent scan ----------------
// grid = (64 j-tiles, 2 m-tiles), block = 96 (3 waves; wave g = gate r/z/n).
// Each WG keeps its 3x16x1024 bf16 W_hh slice resident in LDS for all 512 steps.
// The slice preload uses the Tensor Data Mover (tensor_load_to_lds + s_wait_tensorcnt)
// when the builtin is available; manual vector copy otherwise.
#define SMEM_SCAN (3 * 16 * HH * 2 + 2 * 16 * 16 * 4)

__global__ __launch_bounds__(96) void k_gru_scan(
    const unsigned short* __restrict__ Whh,  // [G3][HH] bf16
    const float* __restrict__ XP,            // [L*B][G3]
    const float* __restrict__ bhh,           // [G3] (layer's b_hh, f32)
    unsigned short* __restrict__ Hseq,       // [(L+1)][B][HH] bf16; slice 0 pre-zeroed
    float* __restrict__ out,                 // last layer: [B][L][HH] f32, else nullptr
    unsigned int* __restrict__ flags,        // [L][2] arrival counters (pre-zeroed)
    int isLast)
{
    extern __shared__ unsigned char smem[];
    unsigned short* lw  = (unsigned short*)smem;                 // [3][16][HH]
    float*          xch = (float*)(smem + 3 * 16 * HH * 2);      // [2][16][16] r,z exchange

    const int lane = threadIdx.x & 31;
    const int g    = threadIdx.x >> 5;   // gate: 0=r 1=z 2=n
    const int jt   = blockIdx.x;         // 0..63 hidden column tile
    const int mt   = blockIdx.y;         // 0..1 batch tile

    // --- Preload this gate's W_hh slice (16 rows x 2048B, row stride 2048B) into LDS once. ---
    {
        const unsigned short* gsrc = Whh + ((size_t)g * HH + (size_t)jt * 16) * HH;
        unsigned short*       ldst = lw + g * 16 * HH;
#if __has_builtin(__builtin_amdgcn_tensor_load_to_lds) && __has_builtin(__builtin_amdgcn_s_wait_tensorcnt)
        // Tensor DMA descriptor (D#): 2D tile, data_size=8B units.
        unsigned long long ga = (unsigned long long)(const void*)gsrc;
        unsigned int lds_off  = (unsigned int)(unsigned long long)(void*)ldst;
        u32x4 d0;
        d0[0] = 0x1u;                                            // count=1, user D#
        d0[1] = lds_off;                                         // lds_addr (bytes)
        d0[2] = (unsigned int)(ga & 0xFFFFFFFFu);                // global_addr[31:0]
        d0[3] = (unsigned int)((ga >> 32) & 0x01FFFFFFu) | (2u << 30); // addr[56:32] | type=2
        i32x8 d1;
        d1[0] = (3 << 16);                 // wg_mask=0, data_size=8B
        d1[1] = (int)(256u << 16);         // tensor_dim0[15:0] = 256 (units of 8B = 2048B row)
        d1[2] = (int)(16u << 16);          // tensor_dim0[31:16]=0 | tensor_dim1[15:0]=16
        d1[3] = (int)(256u << 16);         // tensor_dim1[31:16]=0 | tile_dim0=256
        d1[4] = 16;                        // tile_dim1=16 | tile_dim2=0
        d1[5] = 256;                       // tensor_dim0_stride[31:0] = 256 (2048B)
        d1[6] = 0;                         // stride0[47:32]=0 | stride1[15:0]=0
        d1[7] = 0;
        i32x4 dz4 = {0, 0, 0, 0};
        i32x8 dz8 = {0, 0, 0, 0, 0, 0, 0, 0};
        __builtin_amdgcn_tensor_load_to_lds(d0, d1, dz4, dz4, dz8, 0);
        __builtin_amdgcn_s_wait_tensorcnt(0);
#else
        const uint4* s = (const uint4*)gsrc;
        uint4*       d = (uint4*)ldst;
        for (int i = lane; i < 16 * HH / 8; i += 32) d[i] = s[i];
#endif
    }
    __syncthreads();

    const int   n    = lane & 15;
    const int   mrow = (lane >> 4) * 8;
    const int   gcol = g * HH + jt * 16 + n;          // xp column for this lane
    const float bn   = bhh[2 * HH + jt * 16 + n];     // b_hh_n (multiplied by r)
    const unsigned short* Bw = lw + g * 16 * HH;

    for (int t = 0; t < LL; ++t) {
        if (t > 0) {
            // wait until all 64 j-tiles of our m-tile published h_{t-1}
            if (threadIdx.x == 0) {
                while (atomicAdd(&flags[(t - 1) * 2 + mt], 0u) < 64u)
                    __builtin_amdgcn_s_sleep(1);
            }
            __syncthreads();
            __threadfence();  // acquire published h_{t-1}
        }
        const unsigned short* Hin = Hseq + (size_t)t * BB * HH + (size_t)mt * 16 * HH;

        // hint: next step's xp rows
        int tn = (t + 1 < LL) ? (t + 1) : t;
        __builtin_prefetch(&XP[((size_t)tn * BB + mt * 16) * G3 + gcol], 0, 1);

        v8f acc = {};
        #pragma unroll 4
        for (int kk = 0; kk < HH / 32; ++kk) {
            v16bf a = load_a_frag(Hin + kk * 32, HH, lane);
            v16bf b = load_b_frag(Bw  + kk * 32, HH, lane);
            acc = __builtin_amdgcn_wmma_f32_16x16x32_bf16(false, a, false, b, (short)0, acc, false, false);
        }

        float xp[8];
        #pragma unroll
        for (int v = 0; v < 8; ++v) {
            size_t m = (size_t)t * BB + mt * 16 + mrow + v;
            xp[v] = XP[m * G3 + gcol];
        }

        if (g == 0) {          // r = sigmoid(xr + hr)   (biases folded into xp)
            #pragma unroll
            for (int v = 0; v < 8; ++v) xch[(mrow + v) * 16 + n] = sigmoidf_(acc[v] + xp[v]);
        } else if (g == 1) {   // z = sigmoid(xz + hz)
            #pragma unroll
            for (int v = 0; v < 8; ++v) xch[256 + (mrow + v) * 16 + n] = sigmoidf_(acc[v] + xp[v]);
        }
        __syncthreads();

        if (g == 2) {          // n = tanh(xn + r*(h.Whh_n + b_hh_n)); h' = (1-z)n + z h
            #pragma unroll
            for (int v = 0; v < 8; ++v) {
                int   m  = mrow + v;
                int   bg = mt * 16 + m;
                float r  = xch[m * 16 + n];
                float z  = xch[256 + m * 16 + n];
                float hn = acc[v] + bn;
                float nn = tanhf(xp[v] + r * hn);
                float hp = bf2f(Hseq[(size_t)t * BB * HH + (size_t)bg * HH + jt * 16 + n]);
                float hv = (1.0f - z) * nn + z * hp;
                Hseq[(size_t)(t + 1) * BB * HH + (size_t)bg * HH + jt * 16 + n] = f2bf(hv);
                if (isLast) out[((size_t)bg * LL + t) * HH + jt * 16 + n] = hv;
            }
            __threadfence();   // make h_t visible device-wide before publish
        }
        __syncthreads();       // also protects xch reuse next iteration

        if (threadIdx.x == 0) {
            __threadfence();
            atomicAdd(&flags[t * 2 + mt], 1u);  // publish our j-tile of h_t
        }
    }
}

// ---------------- host orchestration ----------------
extern "C" void kernel_launch(void* const* d_in, const int* in_sizes, int n_in,
                              void* d_out, int out_size, void* d_ws, size_t ws_size,
                              hipStream_t stream) {
    const float* x    = (const float*)d_in[0];   // [B][L][IN]
    const float* Wih0 = (const float*)d_in[1];   // [3H][IN]
    const float* WihR = (const float*)d_in[2];   // [NL-1][3H][H]
    const float* Whh  = (const float*)d_in[3];   // [NL][3H][H]
    const float* bih  = (const float*)d_in[4];   // [NL][3H]
    const float* bhh  = (const float*)d_in[5];   // [NL][3H]
    float* out = (float*)d_out;

    unsigned char* ws = (unsigned char*)d_ws;
    size_t off = 0;
    auto alloc = [&](size_t bytes) -> unsigned char* {
        unsigned char* p = ws + off;
        off += (bytes + 255) & ~(size_t)255;
        return p;
    };

    unsigned short* p_wih0  = (unsigned short*)alloc((size_t)G3 * IN_F * 2);
    unsigned short* p_wihR  = (unsigned short*)alloc((size_t)(NLAY - 1) * G3 * HH * 2);
    unsigned short* p_whh   = (unsigned short*)alloc((size_t)NLAY * G3 * HH * 2);
    float*          p_biasC = (float*)alloc((size_t)NLAY * G3 * 4);
    float*          p_xp    = (float*)alloc((size_t)LL * BB * G3 * 4);
    unsigned short* p_seqA  = (unsigned short*)alloc((size_t)(LL + 1) * BB * HH * 2);
    unsigned short* p_seqB  = (unsigned short*)alloc((size_t)(LL + 1) * BB * HH * 2);
    unsigned int*   p_flags = (unsigned int*)alloc((size_t)2 * LL * 4);

    (void)hipFuncSetAttribute((const void*)k_gru_scan,
                              hipFuncAttributeMaxDynamicSharedMemorySize, SMEM_SCAN);

    k_cast_bf16<<<2048, 256, 0, stream>>>(Wih0, p_wih0, (size_t)G3 * IN_F);
    k_cast_bf16<<<4096, 256, 0, stream>>>(WihR, p_wihR, (size_t)(NLAY - 1) * G3 * HH);
    k_cast_bf16<<<4096, 256, 0, stream>>>(Whh,  p_whh,  (size_t)NLAY * G3 * HH);
    k_biasc<<<(NLAY * G3 + 255) / 256, 256, 0, stream>>>(bih, bhh, p_biasC);
    k_cast_x<<<4096, 256, 0, stream>>>(x, p_seqA);

    unsigned short* Xin  = p_seqA;
    unsigned short* Hout = p_seqB;
    for (int n = 0; n < NLAY; ++n) {
        const int F = (n == 0) ? IN_F : HH;
        const unsigned short* Wih = (n == 0) ? p_wih0 : p_wihR + (size_t)(n - 1) * G3 * HH;
        const int isLast = (n == NLAY - 1) ? 1 : 0;

        dim3 gproj(G3 / 256, (LL * BB) / 32);
        k_xproj<<<gproj, 128, 0, stream>>>(Xin + (size_t)BB * F, Wih, p_biasC + n * G3, p_xp, F);

        k_zero<<<(BB * HH / 2 + 255) / 256, 256, 0, stream>>>(
            p_flags, (unsigned int*)Hout, 2 * LL, BB * HH / 2);

        dim3 gscan(HH / 16, BB / 16);
        k_gru_scan<<<gscan, 96, SMEM_SCAN, stream>>>(
            p_whh + (size_t)n * G3 * HH, p_xp, bhh + n * G3,
            Hout, isLast ? out : nullptr, p_flags, isLast);

        unsigned short* tmp = Xin; Xin = Hout; Hout = tmp;
    }
}
